// LabelwiseAttention_12214886990680
// MI455X (gfx1250) — matrix operand order repaired and checked by reading, not compiled, not run on hardware
//
#include <hip/hip_runtime.h>
#include <hip/hip_bf16.h>

// ---- problem constants ----
#define B_  4
#define L_  2048
#define D_  512
#define C_  8921
#define CT_ 128   // classes per workgroup (8 waves x 16)

typedef __bf16 bf16_t;
typedef __attribute__((ext_vector_type(16))) __bf16 v16bf;
typedef __attribute__((ext_vector_type(8)))  float  v8f;
typedef __attribute__((ext_vector_type(4)))  float  v4f;

union BfFrag {
    v16bf  v;
    v4f    f4[2];
    bf16_t e[16];
};

// ---------- prep: f32 -> bf16 conversions ----------
__global__ void cvt_w_kernel(const float* __restrict__ W, bf16_t* __restrict__ Wb, int n) {
    int i = blockIdx.x * blockDim.x + threadIdx.x;
    if (i < n) Wb[i] = (bf16_t)W[i];
}

// inputs [B,L,D] -> bf16 [B,L,D] and transposed bf16 [B,D,L]
__global__ void cvt_x_kernel(const float* __restrict__ X,
                             bf16_t* __restrict__ Xb,
                             bf16_t* __restrict__ XTb) {
    int i = blockIdx.x * blockDim.x + threadIdx.x;   // 0 .. B*L*D-1
    int b = i >> 20;             // L*D = 2^20
    int r = i & (L_ * D_ - 1);
    int l = r >> 9;              // D = 2^9
    int d = r & (D_ - 1);
    float v = X[i];
    Xb[i] = (bf16_t)v;
    XTb[((size_t)b * D_ + d) * L_ + l] = (bf16_t)v;
}

// ---------- main fused kernel ----------
__global__ __launch_bounds__(256)
void lw_attn_kernel(const bf16_t* __restrict__ Wb,   // [C,D] bf16
                    const bf16_t* __restrict__ Xb,   // [B,L,D] bf16
                    const bf16_t* __restrict__ XTb,  // [B,D,L] bf16
                    const unsigned char* __restrict__ masks, // [B,L] bool
                    float* __restrict__ logits,      // [B,C,D]
                    float* __restrict__ attn)        // [B,C,L]
{
    const int b    = blockIdx.y;
    const int wave = threadIdx.x >> 5;
    const int lane = threadIdx.x & 31;
    const int c0   = blockIdx.x * CT_ + wave * 16;
    if (c0 >= C_) return;

    const int nl = lane & 15;   // N (or M) index within 16
    const int h  = lane >> 4;   // lane half

    const bf16_t* Xrow = Xb  + (size_t)b * L_ * D_;
    const bf16_t* XT   = XTb + (size_t)b * D_ * L_;
    const unsigned char* mk = masks + (size_t)b * L_;
    float* attnB = attn   + (size_t)b * C_ * L_;
    float* logB  = logits + (size_t)b * C_ * D_;

    // ---- resident A fragments: W tile 16 x 512 (ISA 16-bit A layout:
    //      half h holds K = h*8..h*8+7 and 16+h*8..16+h*8+7 per 32-K step) ----
    v16bf afrag[16];
    {
        int crow = c0 + nl; if (crow > C_ - 1) crow = C_ - 1;
        const bf16_t* wp = Wb + (size_t)crow * D_;
#pragma unroll
        for (int k = 0; k < 16; ++k) {
            BfFrag u;
            u.f4[0] = *(const v4f*)(wp + k * 32 + h * 8);
            u.f4[1] = *(const v4f*)(wp + k * 32 + 16 + h * 8);
            afrag[k] = u.v;
        }
    }

    float rmax[8];
#pragma unroll
    for (int j = 0; j < 8; ++j) rmax[j] = -__builtin_inff();

    // ---- pass 1: scores, store raw, track row max ----
    for (int lc = 0; lc < L_; lc += 16) {
        v8f acc = {};
        const bf16_t* xp = Xrow + (size_t)(lc + nl) * D_;
#pragma unroll
        for (int k = 0; k < 16; ++k) {
            BfFrag ub;
            ub.f4[0] = *(const v4f*)(xp + k * 32 + h * 8);
            ub.f4[1] = *(const v4f*)(xp + k * 32 + 16 + h * 8);
            v16bf bfrag = ub.v;
            acc = __builtin_amdgcn_wmma_f32_16x16x32_bf16(
                      false, afrag[k], false, bfrag, (short)0, acc, false, false);
        }
        const bool mval = mk[lc + nl] != 0;
#pragma unroll
        for (int j = 0; j < 8; ++j) {
            float s = mval ? acc[j] : -__builtin_inff();
            rmax[j] = fmaxf(rmax[j], s);
            int cr = c0 + h * 8 + j;
            if (cr < C_) attnB[(size_t)cr * L_ + lc + nl] = s;
        }
    }

    // reduce row max over the 16 lanes of each half (row = h*8 + j)
#pragma unroll
    for (int j = 0; j < 8; ++j) {
#pragma unroll
        for (int m = 1; m < 16; m <<= 1)
            rmax[j] = fmaxf(rmax[j], __shfl_xor(rmax[j], m, 32));
        if (rmax[j] == -__builtin_inff()) rmax[j] = 0.f;
    }

    asm volatile("s_wait_storecnt 0x0" ::: "memory");

    // ---- pass 2: e = exp(s - max), row sums ----
    float rsum[8];
#pragma unroll
    for (int j = 0; j < 8; ++j) rsum[j] = 0.f;
#pragma unroll
    for (int j = 0; j < 8; ++j) {
        int cr = c0 + h * 8 + j;
        if (cr >= C_) continue;
        float* rowp = attnB + (size_t)cr * L_;
        float mx = rmax[j];
        float s = 0.f;
        for (int p = nl * 4; p < L_; p += 64) {
            v4f v = *(const v4f*)(rowp + p);
            v.x = __expf(v.x - mx); v.y = __expf(v.y - mx);
            v.z = __expf(v.z - mx); v.w = __expf(v.w - mx);
            *(v4f*)(rowp + p) = v;
            s += (v.x + v.y) + (v.z + v.w);
        }
        rsum[j] = s;
    }
#pragma unroll
    for (int j = 0; j < 8; ++j) {
#pragma unroll
        for (int m = 1; m < 16; m <<= 1)
            rsum[j] += __shfl_xor(rsum[j], m, 32);
    }

    asm volatile("s_wait_storecnt 0x0" ::: "memory");

    // ---- pass 2b: normalize attention in place ----
#pragma unroll
    for (int j = 0; j < 8; ++j) {
        int cr = c0 + h * 8 + j;
        if (cr >= C_) continue;
        float inv = 1.0f / rsum[j];
        float* rowp = attnB + (size_t)cr * L_;
        for (int p = nl * 4; p < L_; p += 64) {
            v4f v = *(const v4f*)(rowp + p);
            v.x *= inv; v.y *= inv; v.z *= inv; v.w *= inv;
            *(v4f*)(rowp + p) = v;
        }
    }

    asm volatile("s_wait_storecnt 0x0" ::: "memory");

    // ---- pass 3: logits = attn x X  (A = normalized attention, on-the-fly bf16) ----
    int crowA = c0 + nl; if (crowA > C_ - 1) crowA = C_ - 1;
    const float* arow = attnB + (size_t)crowA * L_;

#pragma unroll 1
    for (int ng = 0; ng < 4; ++ng) {          // 4 groups of 8 N-chunks (D = 32 x 16)
        v8f acc3[8];
#pragma unroll
        for (int i = 0; i < 8; ++i) { v8f z = {}; acc3[i] = z; }

        for (int kt = 0; kt < L_; kt += 32) { // K loop over L
            BfFrag ua;
            v4f fs0 = *(const v4f*)(arow + kt + h * 8);
            v4f fs1 = *(const v4f*)(arow + kt + h * 8 + 4);
            v4f fs2 = *(const v4f*)(arow + kt + 16 + h * 8);
            v4f fs3 = *(const v4f*)(arow + kt + 16 + h * 8 + 4);
            v4f fs[4] = {fs0, fs1, fs2, fs3};
#pragma unroll
            for (int q = 0; q < 16; ++q) ua.e[q] = (bf16_t)fs[q >> 2][q & 3];
            v16bf a3 = ua.v;
#pragma unroll
            for (int i = 0; i < 8; ++i) {
                int d0 = (ng * 8 + i) * 16;
                BfFrag ub;
                ub.f4[0] = *(const v4f*)(XT + (size_t)(d0 + nl) * L_ + kt + h * 8);
                ub.f4[1] = *(const v4f*)(XT + (size_t)(d0 + nl) * L_ + kt + 16 + h * 8);
                v16bf b3 = ub.v;
                acc3[i] = __builtin_amdgcn_wmma_f32_16x16x32_bf16(
                              false, a3, false, b3, (short)0, acc3[i], false, false);
            }
        }
#pragma unroll
        for (int i = 0; i < 8; ++i) {
            int d0 = (ng * 8 + i) * 16;
#pragma unroll
            for (int j = 0; j < 8; ++j) {
                int cr = c0 + h * 8 + j;
                if (cr < C_) logB[(size_t)cr * D_ + d0 + nl] = acc3[i][j];
            }
        }
    }
}

// ---------- host launch ----------
extern "C" void kernel_launch(void* const* d_in, const int* in_sizes, int n_in,
                              void* d_out, int out_size, void* d_ws, size_t ws_size,
                              hipStream_t stream) {
    (void)in_sizes; (void)n_in; (void)out_size; (void)ws_size;

    const float*         X     = (const float*)d_in[0];
    const unsigned char* masks = (const unsigned char*)d_in[1];
    const float*         W     = (const float*)d_in[2];

    float* logits = (float*)d_out;                              // [B,C,D]
    float* attn   = (float*)d_out + (size_t)B_ * C_ * D_;       // [B,C,L]

    char*   ws  = (char*)d_ws;
    bf16_t* Xb  = (bf16_t*)(ws);                                // B*L*D bf16
    bf16_t* XTb = (bf16_t*)(ws + (size_t)B_ * L_ * D_ * 2);     // B*D*L bf16
    bf16_t* Wb  = (bf16_t*)(ws + (size_t)B_ * L_ * D_ * 4);     // C*D bf16

    int nx = B_ * L_ * D_;
    cvt_x_kernel<<<nx / 256, 256, 0, stream>>>(X, Xb, XTb);
    int nw = C_ * D_;
    cvt_w_kernel<<<(nw + 255) / 256, 256, 0, stream>>>(W, Wb, nw);

    dim3 grid((C_ + CT_ - 1) / CT_, B_);
    lw_attn_kernel<<<grid, 256, 0, stream>>>(Wb, Xb, XTb, masks, logits, attn);
}
